// ConvolutionModule_32744830665311
// MI455X (gfx1250) — matrix operand order repaired and checked by reading, not compiled, not run on hardware
//
#include <hip/hip_runtime.h>
#include <hip/hip_bf16.h>
#include <math.h>

// ---------------------------------------------------------------------------
// Types for CDNA5 WMMA (wave32): A/B = 16 bf16 (8 VGPRs), C/D = 8 f32.
// ---------------------------------------------------------------------------
typedef __attribute__((ext_vector_type(16))) __bf16 v16bf;
typedef __attribute__((ext_vector_type(8)))  float  v8f;

struct __align__(16) U4 { unsigned int x, y, z, w; };   // POD 16B vector load

union FragAB {
    v16bf        v;
    unsigned int u[8];
    U4           q[2];
};

// f32 -> bf16 (round-to-nearest-even), pack two into one dword.
__device__ __forceinline__ unsigned int pack2_bf16(float a, float b) {
    unsigned int ua = __float_as_uint(a);
    unsigned int ub = __float_as_uint(b);
    ua += 0x7FFFu + ((ua >> 16) & 1u);
    ub += 0x7FFFu + ((ub >> 16) & 1u);
    return (ua >> 16) | (ub & 0xFFFF0000u);
}

// ---------------------------------------------------------------------------
// Utility: zero a float buffer (grid-stride).
// ---------------------------------------------------------------------------
__global__ void zero_f32(float* __restrict__ p, long long n) {
    long long i = (long long)blockIdx.x * blockDim.x + threadIdx.x;
    long long stride = (long long)gridDim.x * blockDim.x;
    for (; i < n; i += stride) p[i] = 0.0f;
}

// ---------------------------------------------------------------------------
// Pack a [K x 256] f32 weight matrix into WMMA bf16 B-fragment order.
// ISA 7.12.2 (16-bit B 32x16): dword v (0..3) holds K = base+2v, base+2v+1,
// dword v (4..7) holds K = base+16+2(v-4), +1; base = 0 for lanes 0-15,
// 8 for lanes 16-31; N = lane & 15.
// Linear index: ((kt*16 + nt)*32 + lane)*8 + v
// ---------------------------------------------------------------------------
__global__ void pack_b_kernel(const float* __restrict__ W,
                              unsigned int* __restrict__ P,
                              int K, int ldW) {
    int t = blockIdx.x * blockDim.x + threadIdx.x;
    int total = (K >> 5) << 12;            // (K/32) * 16 * 32 * 8
    if (t >= total) return;
    int v    = t & 7;
    int lane = (t >> 3) & 31;
    int nt   = (t >> 8) & 15;
    int kt   = t >> 12;
    int n  = nt * 16 + (lane & 15);
    int kb = kt * 32 + ((lane >> 4) << 3) + ((v & 3) << 1) + ((v >> 2) << 4);
    P[t] = pack2_bf16(W[(size_t)kb * ldW + n], W[(size_t)(kb + 1) * ldW + n]);
}

// ---------------------------------------------------------------------------
// Pack a [N x ldA] f32 activation matrix into WMMA bf16 A-fragment order.
// ISA 7.12.2 (16-bit A 16x32): lanes 0-15 hold M=lane, K {0..7, 16..23};
// lanes 16-31 hold M=lane-16, K {8..15, 24..31}.  dword v (0..3) packs
// K = khalf+2v, +2v+1; dword v (4..7) packs K = khalf+16+2(v-4), +...+1.
// Linear index: ((tileM*kTiles + kt)*32 + lane)*8 + v
// ---------------------------------------------------------------------------
__global__ void pack_a_kernel(const float* __restrict__ A,
                              unsigned int* __restrict__ P,
                              int rowTiles, int kTiles, int ldA) {
    int t = blockIdx.x * blockDim.x + threadIdx.x;
    int total = rowTiles * kTiles * 256;   // 32 lanes * 8 dwords per tile
    if (t >= total) return;
    int v     = t & 7;
    int lane  = (t >> 3) & 31;
    int kt    = (t >> 8) % kTiles;
    int tileM = (t >> 8) / kTiles;
    int row   = tileM * 16 + (lane & 15);
    int khalf = (lane >> 4) ? 8 : 0;
    int k     = kt * 32 + khalf + ((v & 3) << 1) + ((v >> 2) << 4);
    const float* ap = A + (size_t)row * ldA + k;
    P[t] = pack2_bf16(ap[0], ap[1]);
}

// ---------------------------------------------------------------------------
// Handcrafted features: column sums of x, then fraction-of-total + log(total).
// ---------------------------------------------------------------------------
__global__ void colsum_kernel(const float* __restrict__ x,
                              float* __restrict__ colsum, int N, int D) {
    int col = blockIdx.x;                   // one block per column
    float s = 0.0f;
    for (int i = threadIdx.x; i < N; i += blockDim.x)
        s += x[(size_t)i * D + col];
    __shared__ float sh[256];
    sh[threadIdx.x] = s;
    __syncthreads();
    for (int o = blockDim.x >> 1; o > 0; o >>= 1) {
        if (threadIdx.x < (unsigned)o) sh[threadIdx.x] += sh[threadIdx.x + o];
        __syncthreads();
    }
    if (threadIdx.x == 0) colsum[col] = sh[0];
}

__global__ void handcrafted_kernel(const float* __restrict__ colsum,
                                   float* __restrict__ out, int D) {
    float total = 0.0f;
    for (int j = 0; j < D; ++j) total += colsum[j];   // D=128, trivial
    int t = threadIdx.x;
    if (t < D) out[t] = colsum[t] / total;
    if (t == 0) out[D] = logf(total);
}

// ---------------------------------------------------------------------------
// Edge aggregation: agg[dst] += h[src].  One thread per (edge, 4 features).
// Gather + f32 global atomics; the 51MB feature matrix is L2-resident.
// ---------------------------------------------------------------------------
__global__ void scatter_add_kernel(const float* __restrict__ h,
                                   const int* __restrict__ ei,
                                   float* __restrict__ agg, int E, int D) {
    long long idx = (long long)blockIdx.x * blockDim.x + threadIdx.x;
    int Dq = D >> 2;
    long long total = (long long)E * Dq;
    if (idx >= total) return;
    int e = (int)(idx / Dq);
    int c = ((int)(idx % Dq)) << 2;
    int s = ei[e];
    int d = ei[E + e];
    const float4 v = *(const float4*)(h + (size_t)s * D + c);
    float* p = agg + (size_t)d * D + c;
    atomicAdd(p + 0, v.x);
    atomicAdd(p + 1, v.y);
    atomicAdd(p + 2, v.z);
    atomicAdd(p + 3, v.w);
}

// ---------------------------------------------------------------------------
// Fused GraphConv GEMM on pre-packed bf16 fragments:
//   out = relu(Aagg @ Brel + Aroot @ Broot + bias)
// All operands are bf16 fragments in WMMA register order; the hot loop is
// pure global_load_b128 + v_wmma_f32_16x16x32_bf16.  One wave owns a
// 16(M) x 64(N) tile: 4 accumulators, A fragment reused 4x per k-step.
// Wave-uniform guard keeps EXEC all-ones for WMMA.
// ---------------------------------------------------------------------------
__global__ void gemm_packed_wmma(const unsigned int* __restrict__ PAagg,
                                 const unsigned int* __restrict__ PAroot,
                                 const unsigned int* __restrict__ Brel,
                                 const unsigned int* __restrict__ Broot,
                                 const float* __restrict__ bias,
                                 float* __restrict__ out,
                                 int rowTiles, int kTiles) {
    const int lane = threadIdx.x & 31;
    const int wave = blockIdx.x * (blockDim.x >> 5) + (threadIdx.x >> 5);
    const int totalWaves = rowTiles * 4;          // 4 column groups of 64
    if (wave >= totalWaves) return;               // wave-uniform exit

    const int tileM  = wave >> 2;
    const int cg     = wave & 3;                  // columns [cg*64, cg*64+64)
    const int mlocal = lane & 15;
    const int hi     = lane >> 4;                 // 0: lanes 0-15, 1: 16-31

    v8f acc[4];
#pragma unroll
    for (int j = 0; j < 4; ++j) acc[j] = (v8f){0, 0, 0, 0, 0, 0, 0, 0};

#pragma unroll
    for (int s = 0; s < 2; ++s) {
        const unsigned int* __restrict__ PA = s ? PAroot : PAagg;
        const unsigned int* __restrict__ B  = s ? Broot  : Brel;
        for (int kt = 0; kt < kTiles; ++kt) {
            const U4* ap = (const U4*)(PA + (((size_t)(tileM * kTiles + kt) * 32 + lane) << 3));
            FragAB a;
            a.q[0] = ap[0];                       // global_load_b128
            a.q[1] = ap[1];                       // global_load_b128
            if (kt + 1 < kTiles)                  // global_prefetch_b8
                __builtin_prefetch((const void*)(ap + 16 * 32 / 4), 0, 1);
#pragma unroll
            for (int j = 0; j < 4; ++j) {
                const int nt = cg * 4 + j;
                const U4* bp = (const U4*)(B + (((size_t)(kt * 16 + nt) * 32 + lane) << 3));
                FragAB b;
                b.q[0] = bp[0];                   // global_load_b128
                b.q[1] = bp[1];                   // global_load_b128
                acc[j] = __builtin_amdgcn_wmma_f32_16x16x32_bf16(
                    false, a.v, false, b.v, (short)0, acc[j], false, false);
            }
        }
    }

    // C/D layout: lanes 0-15 -> N=lane, M=r;  lanes 16-31 -> N=lane-16, M=r+8
#pragma unroll
    for (int j = 0; j < 4; ++j) {
        const int n  = cg * 64 + j * 16 + mlocal;
        const float bv = bias[n];
#pragma unroll
        for (int r = 0; r < 8; ++r) {
            const int m = tileM * 16 + r + (hi << 3);
            float v = acc[j][r] + bv;
            out[(size_t)m * 256 + n] = fmaxf(v, 0.0f);
        }
    }
}

// ---------------------------------------------------------------------------
// Global mean pool over batch ids (batch is sorted, NUM_GRAPHS=64).
// ---------------------------------------------------------------------------
__global__ void pool_accum_kernel(const float* __restrict__ h,
                                  const int* __restrict__ batch,
                                  float* __restrict__ pooled,
                                  float* __restrict__ counts, int N) {
    long long idx = (long long)blockIdx.x * blockDim.x + threadIdx.x;
    if (idx >= (long long)N * 256) return;
    int n = (int)(idx >> 8);
    int c = (int)(idx & 255);
    int g = batch[n];
    atomicAdd(&pooled[g * 256 + c], h[(size_t)n * 256 + c]);
    if (c == 0) atomicAdd(&counts[g], 1.0f);
}

__global__ void pool_div_kernel(float* __restrict__ pooled,
                                const float* __restrict__ counts) {
    int idx = blockIdx.x * blockDim.x + threadIdx.x;
    if (idx >= 64 * 256) return;
    pooled[idx] = pooled[idx] / fmaxf(counts[idx >> 8], 1.0f);
}

// ---------------------------------------------------------------------------
// Host-side orchestration (all launches on `stream`, no allocation).
// ---------------------------------------------------------------------------
extern "C" void kernel_launch(void* const* d_in, const int* in_sizes, int n_in,
                              void* d_out, int out_size, void* d_ws, size_t ws_size,
                              hipStream_t stream) {
    const float* x       = (const float*)d_in[0];
    const int*   ei      = (const int*)  d_in[1];   // edge_index [2,E]
    const int*   batch   = (const int*)  d_in[2];
    const float* W1_rel  = (const float*)d_in[3];
    const float* W1_root = (const float*)d_in[4];
    const float* b1      = (const float*)d_in[5];
    const float* W2_rel  = (const float*)d_in[6];
    const float* W2_root = (const float*)d_in[7];
    const float* b2      = (const float*)d_in[8];

    const int N   = in_sizes[2];          // 50000 (multiple of 16)
    const int E   = in_sizes[1] / 2;      // 800000
    const int Din = in_sizes[0] / N;      // 128
    const int Dh  = in_sizes[5];          // 256
    const int G   = 64;                   // NUM_GRAPHS

    // Workspace layout (~154 MB):
    //   agg  [N*Dh] f32   scatter-add accumulator
    //   h    [N*Dh] f32   activations (updated in place each layer)
    //   PA   [N*Dh/2] u32 packed bf16 A-fragments (aggregate operand)
    //   PR   [N*Dh/2] u32 packed bf16 A-fragments (root operand)
    //   colsum/counts, packed bf16 weights (~0.4 MB)
    float* agg    = (float*)d_ws;
    float* h      = agg + (size_t)N * Dh;
    unsigned int* PA = (unsigned int*)(h + (size_t)N * Dh);
    unsigned int* PR = PA + (size_t)N * Dh / 2;
    float* colsum = (float*)(PR + (size_t)N * Dh / 2);
    float* counts = colsum + Din;
    unsigned int* pk1rel  = (unsigned int*)(counts + G);
    unsigned int* pk1root = pk1rel  + (size_t)Din * Dh / 2;
    unsigned int* pk2rel  = pk1root + (size_t)Din * Dh / 2;
    unsigned int* pk2root = pk2rel  + (size_t)Dh  * Dh / 2;

    const int rowTiles = N / 16;
    const int kT1 = Din / 32, kT2 = Dh / 32;
    const int waves    = rowTiles * 4;
    const int gblocks  = (waves + 7) / 8;          // 8 waves / 256-thread block
    const int aTot1 = rowTiles * kT1 * 256;        // pack_a thread counts
    const int aTot2 = rowTiles * kT2 * 256;

    // 1) Pack weights into WMMA fragment order (small, once per launch).
    {
        int tot1 = Din * Dh / 2, tot2 = Dh * Dh / 2;
        pack_b_kernel<<<(tot1 + 255) / 256, 256, 0, stream>>>(W1_rel,  pk1rel,  Din, Dh);
        pack_b_kernel<<<(tot1 + 255) / 256, 256, 0, stream>>>(W1_root, pk1root, Din, Dh);
        pack_b_kernel<<<(tot2 + 255) / 256, 256, 0, stream>>>(W2_rel,  pk2rel,  Dh,  Dh);
        pack_b_kernel<<<(tot2 + 255) / 256, 256, 0, stream>>>(W2_root, pk2root, Dh,  Dh);
    }

    // 2) Handcrafted features -> d_out[G*Dh .. G*Dh+Din]
    colsum_kernel<<<Din, 256, 0, stream>>>(x, colsum, N, Din);
    handcrafted_kernel<<<1, 256, 0, stream>>>(colsum, (float*)d_out + (size_t)G * Dh, Din);

    // 3) Layer 1: agg(x); h = relu(agg@W1_rel + x@W1_root + b1)
    zero_f32<<<4096, 256, 0, stream>>>(agg, (long long)N * Din);
    {
        long long t = (long long)E * (Din / 4);
        scatter_add_kernel<<<(int)((t + 255) / 256), 256, 0, stream>>>(x, ei, agg, E, Din);
    }
    pack_a_kernel<<<(aTot1 + 255) / 256, 256, 0, stream>>>(agg, PA, rowTiles, kT1, Din);
    pack_a_kernel<<<(aTot1 + 255) / 256, 256, 0, stream>>>(x,   PR, rowTiles, kT1, Din);
    gemm_packed_wmma<<<gblocks, 256, 0, stream>>>(PA, PR, pk1rel, pk1root, b1, h,
                                                  rowTiles, kT1);

    // 4) Layers 2 and 3 (shared weights): agg(h); h = relu(...) in place.
    for (int layer = 0; layer < 2; ++layer) {
        zero_f32<<<4096, 256, 0, stream>>>(agg, (long long)N * Dh);
        long long t = (long long)E * (Dh / 4);
        scatter_add_kernel<<<(int)((t + 255) / 256), 256, 0, stream>>>(h, ei, agg, E, Dh);
        pack_a_kernel<<<(aTot2 + 255) / 256, 256, 0, stream>>>(agg, PA, rowTiles, kT2, Dh);
        pack_a_kernel<<<(aTot2 + 255) / 256, 256, 0, stream>>>(h,   PR, rowTiles, kT2, Dh);
        gemm_packed_wmma<<<gblocks, 256, 0, stream>>>(PA, PR, pk2rel, pk2root, b2, h,
                                                      rowTiles, kT2);
    }

    // 5) Global mean pool -> d_out[0 .. G*Dh)
    zero_f32<<<64, 256, 0, stream>>>((float*)d_out, (long long)G * Dh);
    zero_f32<<<1, 64, 0, stream>>>(counts, G);
    {
        long long t = (long long)N * 256;
        pool_accum_kernel<<<(int)((t + 255) / 256), 256, 0, stream>>>(h, batch,
                                                                      (float*)d_out, counts, N);
    }
    pool_div_kernel<<<(G * Dh + 255) / 256, 256, 0, stream>>>((float*)d_out, counts);
}
